// RingDilatedAttentionFixed_68959994905214
// MI455X (gfx1250) — matrix operand order repaired and checked by reading, not compiled, not run on hardware
//
#include <hip/hip_runtime.h>

typedef __attribute__((ext_vector_type(16))) _Float16 v16h;
typedef __attribute__((ext_vector_type(8)))  _Float16 v8h;
typedef __attribute__((ext_vector_type(4)))  _Float16 v4h;
typedef __attribute__((ext_vector_type(8)))  float    v8f;
typedef __attribute__((ext_vector_type(4)))  float    v4f;

#define N_TOK     7168
#define HEADS     8
#define DDIM      64
#define TOKSTRIDE 512                       // floats between consecutive tokens
#define VT_W      12544                     // 7168 + 3584 + 1792 (dil 1/2/4 copies)
#define K16_ELEMS (N_TOK * TOKSTRIDE)       // 3,670,016
#define K16_BYTES (K16_ELEMS * 2)           // 7,340,032
// total workspace: K16_BYTES + 512*VT_W*2 = 20,185,088 bytes (~20 MB)

// ---------------- pre-pass 1: K fp32 -> fp16, same layout ----------------
__global__ __launch_bounds__(256) void cvt_k_f16(const float* __restrict__ k,
                                                 _Float16* __restrict__ k16) {
  const int i = (blockIdx.x * 256 + threadIdx.x) * 4;
  v4f f = *(const v4f*)(k + i);
  v4h h;
  h[0] = (_Float16)f[0]; h[1] = (_Float16)f[1];
  h[2] = (_Float16)f[2]; h[3] = (_Float16)f[3];
  *(v4h*)(k16 + i) = h;
}

// ------- pre-pass 2: V fp32 -> fp16, transposed, per-dilation copies ------
// vt[(head*64+d) * VT_W + coff + j] = v[(j*dil)*512 + head*64 + d]
// coff = 0 (dil 1, 7168 cols), 7168 (dil 2, 3584 cols), 10752 (dil 4, 1792)
__global__ __launch_bounds__(256) void transpose_v_f16(const float* __restrict__ v,
                                                       _Float16* __restrict__ vt) {
  __shared__ _Float16 tile[64][66];         // [hd][token], padded (33 dwords/row)
  const int bx = blockIdx.x;
  int dil, jbase, coff;
  if (bx < 112)      { dil = 1; jbase = bx * 64;         coff = 0; }
  else if (bx < 168) { dil = 2; jbase = (bx - 112) * 64; coff = 7168; }
  else               { dil = 4; jbase = (bx - 168) * 64; coff = 10752; }
  const int hd0 = blockIdx.y * 64;
  const int tx  = threadIdx.x & 63;         // hd within tile
  const int tr  = threadIdx.x >> 6;         // 0..3
#pragma unroll
  for (int it = 0; it < 16; ++it) {
    const int j = tr + it * 4;              // token-col within tile
    const int token = (jbase + j) * dil;
    tile[tx][j] = (_Float16)v[(size_t)token * TOKSTRIDE + hd0 + tx];
  }
  __syncthreads();
  const int h  = threadIdx.x >> 2;          // hd row 0..63
  const int t0 = (threadIdx.x & 3) * 16;    // 16 consecutive cols per thread
  v8h a, b;
#pragma unroll
  for (int x = 0; x < 8; ++x) { a[x] = tile[h][t0 + x]; b[x] = tile[h][t0 + 8 + x]; }
  _Float16* dst = vt + (size_t)(hd0 + h) * VT_W + coff + jbase + t0;
  *(v8h*)dst       = a;
  *(v8h*)(dst + 8) = b;
}

// ------------------------------ main kernel ------------------------------
__device__ __forceinline__ v8f wmma_f16(v16h a, v16h b, v8f c) {
  return __builtin_amdgcn_wmma_f32_16x16x32_f16(false, a, false, b,
                                                (short)0, c, false, false);
}
__device__ __forceinline__ v16h cat8(v8h lo, v8h hi) {
  return __builtin_shufflevector(lo, hi, 0, 1, 2, 3, 4, 5, 6, 7,
                                 8, 9, 10, 11, 12, 13, 14, 15);
}

// One wave handles TWO 16-row q-tiles (32 q rows) for one head.
// S^T = K * Q^T so probabilities land directly in the PV A-matrix layout.
__global__ __launch_bounds__(256) void dilated_attn_wmma(
    const float* __restrict__ q, const _Float16* __restrict__ k16,
    const _Float16* __restrict__ vt, float* __restrict__ out) {

  const int wave  = (blockIdx.x << 3) | (threadIdx.x >> 5);  // 0..1791
  const int lane  = threadIdx.x & 31;
  const int head  = wave & 7;
  const int stile = wave >> 3;                               // 0..223
  const int g     = lane >> 4;
  const int ln    = lane & 15;

  int dil, segoff;
  if (stile < 32)      { dil = 1; segoff = 0; }      // q [0,1024)
  else if (stile < 96) { dil = 2; segoff = 7168; }   // q [1024,3072)
  else                 { dil = 4; segoff = 10752; }  // q [3072,7168)
  const int kvLen   = N_TOK / dil;
  const int kstride = dil * TOKSTRIDE;
  const int q0      = stile << 5;

  const float QSCALE = 0.125f * 1.44269504088896340736f;  // 1/sqrt(64)*log2(e)

  // ---- Q^T B-regs for both q-tiles (only f32->f16 converts in the kernel)
  v16h bq[2][2];
#pragma unroll
  for (int u = 0; u < 2; ++u) {
    const float* qrow = q + (size_t)(q0 + u * 16 + ln) * TOKSTRIDE + head * DDIM;
#pragma unroll
    for (int j = 0; j < 2; ++j)
#pragma unroll
      for (int p = 0; p < 4; ++p) {
        v4f f = *(const v4f*)(qrow + j * 32 + 8 * g + 4 * p + 8 * (p >> 1));
#pragma unroll
        for (int x = 0; x < 4; ++x) bq[u][j][4 * p + x] = (_Float16)(f[x] * QSCALE);
      }
  }

  v8f acc[2][4];
#pragma unroll
  for (int u = 0; u < 2; ++u)
#pragma unroll
    for (int t = 0; t < 4; ++t) acc[u][t] = (v8f){};
  float m_run[2] = {-1e30f, -1e30f};
  float l_run[2] = {0.0f, 0.0f};
  const v8f zero = (v8f){};

  const _Float16* kbase = k16 + head * DDIM;
  const _Float16* vbase = vt + (size_t)(head * DDIM + ln) * VT_W + segoff;

  const int nchunks = kvLen >> 5;  // 224 / 112 / 56
  for (int c = 0; c < nchunks; ++c) {
    const int c0 = c << 5;

    // ---- K chunk (f16, row-major): 2 b128 loads per A-reg, no converts
    v8f st[2][2];
#pragma unroll
    for (int i = 0; i < 2; ++i) {
      const _Float16* krow = kbase + (size_t)(c0 + i * 16 + ln) * kstride;
      v16h a0 = cat8(*(const v8h*)(krow + 8 * g),
                     *(const v8h*)(krow + 8 * g + 16));
      v16h a1 = cat8(*(const v8h*)(krow + 32 + 8 * g),
                     *(const v8h*)(krow + 32 + 8 * g + 16));
#pragma unroll
      for (int u = 0; u < 2; ++u) {
        v8f s    = wmma_f16(a0, bq[u][0], zero);
        st[u][i] = wmma_f16(a1, bq[u][1], s);
      }
    }

    // ---- online softmax (state keyed by q column = ln)
    float sc[2] = {1.0f, 1.0f};
    float mnew[2], rs[2];
    v16h ap[2];
#pragma unroll
    for (int u = 0; u < 2; ++u) {
      float cm = -1e30f;
#pragma unroll
      for (int r = 0; r < 8; ++r) cm = fmaxf(cm, fmaxf(st[u][0][r], st[u][1][r]));
      cm      = fmaxf(cm, __shfl_xor(cm, 16));
      mnew[u] = fmaxf(m_run[u], cm);
      float s = 0.0f;
#pragma unroll
      for (int r = 0; r < 8; ++r) {
        const float e0 = __builtin_amdgcn_exp2f(st[u][0][r] - mnew[u]);
        const float e1 = __builtin_amdgcn_exp2f(st[u][1][r] - mnew[u]);
        s += e0 + e1;
        ap[u][r]     = (_Float16)e0;   // P drops straight into A layout
        ap[u][8 + r] = (_Float16)e1;
      }
      s += __shfl_xor(s, 16);
      rs[u] = s;
    }

    // ---- rescale O only when some row max actually increased (rare)
    const int up = (mnew[0] > m_run[0]) | (mnew[1] > m_run[1]);
    if (__any(up)) {
#pragma unroll
      for (int u = 0; u < 2; ++u) {
        sc[u] = __builtin_amdgcn_exp2f(m_run[u] - mnew[u]);
        float scr[8];
#pragma unroll
        for (int r = 0; r < 8; ++r) scr[r] = __shfl(sc[u], r + 8 * g);
#pragma unroll
        for (int t = 0; t < 4; ++t)
#pragma unroll
          for (int r = 0; r < 8; ++r) acc[u][t][r] *= scr[r];
      }
    }
#pragma unroll
    for (int u = 0; u < 2; ++u) {
      l_run[u] = l_run[u] * sc[u] + rs[u];
      m_run[u] = mnew[u];
    }

    // ---- O += P * V : V^T (f16, kv-contiguous) -> 2 b128 loads per B-reg
    const _Float16* vrow = vbase + c0 + 8 * g;
#pragma unroll
    for (int t = 0; t < 4; ++t) {
      const _Float16* p = vrow + (size_t)(t * 16) * VT_W;
      v16h bv = cat8(*(const v8h*)p, *(const v8h*)(p + 16));
      acc[0][t] = wmma_f16(ap[0], bv, acc[0][t]);
      acc[1][t] = wmma_f16(ap[1], bv, acc[1][t]);
    }
  }

  // ---- finalize
#pragma unroll
  for (int u = 0; u < 2; ++u)
#pragma unroll
    for (int r = 0; r < 8; ++r) {
      const float lr  = __shfl(l_run[u], r + 8 * g);
      const float inv = 1.0f / lr;
      float* orow = out + (size_t)(q0 + u * 16 + r + 8 * g) * TOKSTRIDE + head * DDIM;
#pragma unroll
      for (int t = 0; t < 4; ++t) orow[t * 16 + ln] = acc[u][t][r] * inv;
    }
}

extern "C" void kernel_launch(void* const* d_in, const int* in_sizes, int n_in,
                              void* d_out, int out_size, void* d_ws,
                              size_t ws_size, hipStream_t stream) {
  const float* q = (const float*)d_in[0];
  const float* k = (const float*)d_in[1];
  const float* v = (const float*)d_in[2];
  float* out = (float*)d_out;
  (void)in_sizes; (void)n_in; (void)out_size; (void)ws_size;

  _Float16* k16 = (_Float16*)d_ws;
  _Float16* vt  = (_Float16*)((char*)d_ws + K16_BYTES);

  // pre-pass: fp32 -> fp16 K copy and per-dilation transposed V copies
  hipLaunchKernelGGL(cvt_k_f16, dim3(K16_ELEMS / 1024), dim3(256), 0, stream, k, k16);
  hipLaunchKernelGGL(transpose_v_f16, dim3(196, 8), dim3(256), 0, stream, v, vt);

  // 224 super-tiles (32 q rows) * 8 heads = 1792 waves; 8 waves per block
  hipLaunchKernelGGL(dilated_attn_wmma, dim3(224), dim3(256), 0, stream,
                     q, k16, vt, out);
}
